// ScaleDotAtt_49898930045029
// MI455X (gfx1250) — matrix operand order repaired and checked by reading, not compiled, not run on hardware
//
#include <hip/hip_runtime.h>

// ---------------------------------------------------------------------------
// MI455X (gfx1250) fused VAE-attention forward.
// - f16 WMMA (16x16x32, f32 accum) for every GEMM
// - all activations live in a per-wave LDS slab
// - weights pre-transposed/converted to f16 in ws
// - x pre-converted to padded f16 in ws; per-wave tile DMA'd into LDS by the
//   Tensor Data Mover (tensor_load_to_lds + s_wait_tensorcnt)
// ---------------------------------------------------------------------------

typedef __attribute__((ext_vector_type(16))) _Float16     v16h;
typedef __attribute__((ext_vector_type(8)))  float        v8f;
typedef __attribute__((ext_vector_type(4))) unsigned int  v4u;
typedef __attribute__((ext_vector_type(8)))  int          v8i;
typedef __attribute__((ext_vector_type(4)))  int          v4i;

#if defined(__has_builtin)
#if __has_builtin(__builtin_amdgcn_tensor_load_to_lds)
#define HAVE_TDM 1
#endif
#endif
#ifndef HAVE_TDM
#define HAVE_TDM 0
#endif

constexpr int B_   = 16384;
constexpr int OBS  = 568;
constexpr int H    = 256;
constexpr int SD   = 36;    // SELF_DIM
constexpr int OD   = 28;    // OTHER_DIM
constexpr int NA   = 19;    // N_OTHER
constexpr int KX   = 576;   // OBS padded to multiple of 32
constexpr int NC   = H / 16;

// f16 weight workspace layout (element offsets), all [N=256][Kpad] transposed
constexpr size_t SZ_MU    = (size_t)H * KX;
constexpr size_t OFF_MU   = 0;
constexpr size_t OFF_VAR  = OFF_MU   + SZ_MU;
constexpr size_t OFF_SELF = OFF_VAR  + SZ_MU;          // [256][64]
constexpr size_t OFF_OTH  = OFF_SELF + (size_t)H * 64; // [256][32]
constexpr size_t OFF_Q    = OFF_OTH  + (size_t)H * 32; // [256][256]
constexpr size_t OFF_K    = OFF_Q    + (size_t)H * H;
constexpr size_t OFF_V    = OFF_K    + (size_t)H * H;  // [256][512]
constexpr size_t OFF_A    = OFF_V    + (size_t)H * 2 * H;
constexpr size_t OFF_X16  = OFF_A    + (size_t)H * 2 * H; // [B][576] f16 x
constexpr size_t WS_HALVES_NEEDED = OFF_X16 + (size_t)B_ * KX; // ~20.3 MB

// Per-wave LDS slab: halves 16*576 + 4*16*256 = 25600 (51200 B),
// floats 16*256 + 16*20 + 16*20 + 16*16 = 4992 (19968 B) => 71168 B
constexpr int SLAB_BYTES = 71168;

// --- Transpose + f16-convert weight: dst[n*Kpad + k] = W[k][n], zero-padded K
__global__ void prep_w_kernel(const float* __restrict__ src,
                              _Float16* __restrict__ dst, int Korig, int Kpad) {
  int idx = blockIdx.x * blockDim.x + threadIdx.x;
  if (idx >= H * Kpad) return;
  int n = idx / Kpad, k = idx - n * Kpad;
  dst[idx] = (k < Korig) ? (_Float16)src[(size_t)k * H + n] : (_Float16)0.f;
}

// --- Convert x to f16, row-padded 568 -> 576 (zeros), row-major [B][576]
__global__ void prep_x_kernel(const float* __restrict__ src,
                              _Float16* __restrict__ dst) {
  size_t idx = (size_t)blockIdx.x * blockDim.x + threadIdx.x;
  if (idx >= (size_t)B_ * KX) return;
  size_t row = idx / KX;
  int c = (int)(idx - row * KX);
  dst[idx] = (c < OBS) ? (_Float16)src[row * OBS + c] : (_Float16)0.f;
}

// A fragment: rows in LDS row-major [16][kstride] f16; kelem = starting K (even).
// ISA layout: lanes 0-15 hold K koff..koff+7 and koff+16..+23 (koff=0),
// lanes 16-31 use koff=8.  kelem may be only 4B-aligned -> dword loads.
__device__ __forceinline__ v16h load_a_lds(const _Float16* base, int kstride,
                                           int kelem) {
  const int lane = threadIdx.x & 31;
  const int row  = lane & 15;
  const int koff = (lane >> 4) * 8;
  const _Float16* p = base + row * kstride + kelem + koff;
  v16h a;
  unsigned int* ai = (unsigned int*)&a;
#pragma unroll
  for (int i = 0; i < 4; ++i) ai[i] = *(const unsigned int*)(p + 2 * i);
#pragma unroll
  for (int i = 0; i < 4; ++i) ai[4 + i] = *(const unsigned int*)(p + 16 + 2 * i);
  return a;
}

// B fragment from transposed f16 weights [N][K] in global memory.
// ISA layout: lanes 0-15 hold K kc..kc+7 and kc+16..+23 of column n, lanes 16-31 K kc+8..+15 and kc+24..+31.
__device__ __forceinline__ v16h load_b_w(const _Float16* wt, int K, int ncol0,
                                         int kc) {
  const int lane = threadIdx.x & 31;
  const int col  = ncol0 + (lane & 15);
  const int k0   = kc + (lane >> 4) * 8;
  const _Float16* p = wt + (size_t)col * K + k0;
  v16h b;
  ((v4u*)&b)[0] = *(const v4u*)p;
  ((v4u*)&b)[1] = *(const v4u*)(p + 16);
  return b;
}

__device__ __forceinline__ v8f wmma16(v16h a, v16h b, v8f c) {
  return __builtin_amdgcn_wmma_f32_16x16x32_f16(false, a, false, b, (short)0, c,
                                                false, false);
}

__global__ __launch_bounds__(64) void fused_kernel(
    const float* __restrict__ x, const float* __restrict__ eps,
    const float* __restrict__ b_mu, const float* __restrict__ b_var,
    const float* __restrict__ b_self, const float* __restrict__ b_oth,
    const float* __restrict__ b_q, const float* __restrict__ b_k,
    const float* __restrict__ b_v, const float* __restrict__ b_a,
    const _Float16* __restrict__ wmu, const _Float16* __restrict__ wvar,
    const _Float16* __restrict__ wself, const _Float16* __restrict__ woth,
    const _Float16* __restrict__ wq, const _Float16* __restrict__ wk,
    const _Float16* __restrict__ wv, const _Float16* __restrict__ wa,
    const _Float16* __restrict__ xf16, int use_tdm,
    float* __restrict__ out) {
  extern __shared__ char smem_raw[];
  const int lane = threadIdx.x & 31;
  const int wave = threadIdx.x >> 5;
  const int row0 = (blockIdx.x * 2 + wave) * 16;

  char* slab = smem_raw + (size_t)wave * SLAB_BYTES;
  _Float16* sx   = (_Float16*)slab;   // [16][576] padded x tile
  _Float16* sz   = sx  + 16 * KX;     // [16][256] z
  _Float16* sq   = sz  + 16 * H;      // [16][256] q
  _Float16* ssf  = sq  + 16 * H;      // [16][256] self_em
  _Float16* soem = ssf + 16 * H;      // [16][256] other_em_j / h(f16)
  float* sh   = (float*)(soem + 16 * H); // [16][256] h accumulator
  float* ssc  = sh  + 16 * H;            // [16][20]  scores
  float* satt = ssc + 16 * 20;           // [16][20]  softmax weights
  float* sred = satt + 16 * 20;          // [16][16]  dot-product reduce

  const int n15 = lane & 15;
  const int rb  = (lane >> 4) * 8;  // C-fragment row base

  // ---- stage x tile to LDS as f16 ----
#if HAVE_TDM
  if (use_tdm) {
    // One TDM descriptor per wave: contiguous 16*576 f16 = 18432 B tile,
    // expressed as 2304 x 8-byte elements, DMA'd global -> this wave's slab.
    const unsigned wave_u  = (unsigned)__builtin_amdgcn_readfirstlane(wave);
    const unsigned lds_off = wave_u * (unsigned)SLAB_BYTES;  // sx at slab base
    const unsigned long long ga =
        (unsigned long long)(const void*)xf16 +
        ((unsigned long long)blockIdx.x * 2u + wave_u) * (16ull * KX * 2ull);
    const unsigned elems = 16u * KX * 2u / 8u;  // 2304 units of 8 B

    v4u g0;
    g0[0] = 1u;                                   // count=1, user load
    g0[1] = lds_off;                              // lds_addr (bytes)
    g0[2] = (unsigned)(ga & 0xffffffffull);       // global_addr[31:0]
    g0[3] = (unsigned)((ga >> 32) & 0x01ffffffu)  // global_addr[56:32]
            | (2u << 30);                         // type = 2 ("image")
    v8i g1;
    g1[0] = (int)(3u << 16);       // workgroup_mask=0, data_size=3 (8B)
    g1[1] = (int)(elems << 16);    // tensor_dim0[15:0]  (bits 63:48)
    g1[2] = (int)(1u << 16);       // tensor_dim0 hi=0 | tensor_dim1=1
    g1[3] = (int)(elems << 16);    // tensor_dim1 hi=0 | tile_dim0=2304
    g1[4] = 1;                     // tile_dim1=1, tile_dim2=0
    g1[5] = (int)elems;            // tensor_dim0_stride[31:0]
    g1[6] = 0;                     // stride hi / tensor_dim1_stride lo
    g1[7] = 0;
    v4i gz4 = {0, 0, 0, 0};              // groups 2/3: NULL (<=2D tensor)
    v8i gz8 = {0, 0, 0, 0, 0, 0, 0, 0};  // trailing group (unused)
    __builtin_amdgcn_tensor_load_to_lds(g0, g1, gz4, gz4, gz8, 0);
    __builtin_amdgcn_s_wait_tensorcnt(0);
  } else
#endif
  {
    for (int r = 0; r < 16; ++r) {
      const float* xr = x + (size_t)(row0 + r) * OBS;
      for (int c = lane; c < KX; c += 32)
        sx[r * KX + c] = (c < OBS) ? (_Float16)xr[c] : (_Float16)0.f;
    }
  }
  for (int i = lane; i < 16 * H; i += 32) sh[i] = 0.f;
  __syncthreads();

  // ---- mu / log_var -> z (reparameterize) ----
  for (int nc = 0; nc < NC; ++nc) {
    v8f amu = {}, alv = {};
    for (int kc = 0; kc < KX / 32; ++kc) {
      v16h a  = load_a_lds(sx, KX, kc * 32);
      v16h bm = load_b_w(wmu, KX, nc * 16, kc * 32);
      amu = wmma16(a, bm, amu);
      v16h bv = load_b_w(wvar, KX, nc * 16, kc * 32);
      alv = wmma16(a, bv, alv);
    }
    const int col = nc * 16 + n15;
    const float bm = b_mu[col], bv = b_var[col];
#pragma unroll
    for (int r = 0; r < 8; ++r) {
      const int row = rb + r;
      const float e  = eps[(size_t)(row0 + row) * H + col];
      const float zv = e * __expf(0.5f * (alv[r] + bv)) + (amu[r] + bm);
      sz[row * H + col] = (_Float16)zv;
    }
  }

  // ---- self_em = relu(x[:, :36] @ W_self + b)  (Wt zero-padded K 36..63) ----
  for (int nc = 0; nc < NC; ++nc) {
    v8f acc = {};
    for (int kc = 0; kc < 2; ++kc) {
      v16h a = load_a_lds(sx, KX, kc * 32);
      v16h b = load_b_w(wself, 64, nc * 16, kc * 32);
      acc = wmma16(a, b, acc);
    }
    const int col = nc * 16 + n15;
    const float bb = b_self[col];
#pragma unroll
    for (int r = 0; r < 8; ++r)
      ssf[(rb + r) * H + col] = (_Float16)fmaxf(acc[r] + bb, 0.f);
  }
  __syncthreads();

  // ---- q = relu(z @ W_q + b_q) ----
  for (int nc = 0; nc < NC; ++nc) {
    v8f acc = {};
    for (int kc = 0; kc < H / 32; ++kc) {
      v16h a = load_a_lds(sz, H, kc * 32);
      v16h b = load_b_w(wq, H, nc * 16, kc * 32);
      acc = wmma16(a, b, acc);
    }
    const int col = nc * 16 + n15;
    const float bb = b_q[col];
#pragma unroll
    for (int r = 0; r < 8; ++r)
      sq[(rb + r) * H + col] = (_Float16)fmaxf(acc[r] + bb, 0.f);
  }
  __syncthreads();

  // ---- pass 1: per-agent k_j and scores ----
  for (int j = 0; j < NA; ++j) {
    // other_em_j = relu(x[:, 36+28j : +28] @ W_other + b) (Wt zero rows 28..31)
    for (int nc = 0; nc < NC; ++nc) {
      v16h a = load_a_lds(sx, KX, SD + OD * j);
      v16h b = load_b_w(woth, 32, nc * 16, 0);
      v8f acc = {};
      acc = wmma16(a, b, acc);
      const int col = nc * 16 + n15;
      const float bb = b_oth[col];
#pragma unroll
      for (int r = 0; r < 8; ++r)
        soem[(rb + r) * H + col] = (_Float16)fmaxf(acc[r] + bb, 0.f);
    }
    __syncthreads();
    // k_j = relu(oem @ W_k + b_k); partial row-wise dot with q
    float part[8];
#pragma unroll
    for (int r = 0; r < 8; ++r) part[r] = 0.f;
    for (int nc = 0; nc < NC; ++nc) {
      v8f acc = {};
      for (int kc = 0; kc < H / 32; ++kc) {
        v16h a = load_a_lds(soem, H, kc * 32);
        v16h b = load_b_w(wk, H, nc * 16, kc * 32);
        acc = wmma16(a, b, acc);
      }
      const int col = nc * 16 + n15;
      const float bb = b_k[col];
#pragma unroll
      for (int r = 0; r < 8; ++r) {
        const float kv = fmaxf(acc[r] + bb, 0.f);
        part[r] += kv * (float)sq[(rb + r) * H + col];
      }
    }
#pragma unroll
    for (int r = 0; r < 8; ++r) sred[(rb + r) * 16 + n15] = part[r];
    __syncthreads();
    if (lane < 16) {
      float s = 0.f;
#pragma unroll
      for (int c = 0; c < 16; ++c) s += sred[lane * 16 + c];
      ssc[lane * 20 + j] = s * 0.0625f;  // 1/sqrt(256)
    }
    __syncthreads();
  }

  // ---- softmax over agents; emit att_weight ----
  if (lane < 16) {
    const int row = lane;
    float m = -1e30f;
    for (int j = 0; j < NA; ++j) m = fmaxf(m, ssc[row * 20 + j]);
    float sum = 0.f;
    for (int j = 0; j < NA; ++j) {
      float e = __expf(ssc[row * 20 + j] - m);
      satt[row * 20 + j] = e;
      sum += e;
    }
    const float inv = 1.f / sum;
    for (int j = 0; j < NA; ++j) {
      const float w = satt[row * 20 + j] * inv;
      satt[row * 20 + j] = w;
      out[(size_t)B_ * H + (size_t)(row0 + row) * NA + j] = w;
    }
  }
  __syncthreads();

  // ---- pass 2: v_j = relu([oem_j, self_em] @ W_v + b_v); h += att * v ----
  for (int j = 0; j < NA; ++j) {
    for (int nc = 0; nc < NC; ++nc) {  // recompute other_em_j (cheap)
      v16h a = load_a_lds(sx, KX, SD + OD * j);
      v16h b = load_b_w(woth, 32, nc * 16, 0);
      v8f acc = {};
      acc = wmma16(a, b, acc);
      const int col = nc * 16 + n15;
      const float bb = b_oth[col];
#pragma unroll
      for (int r = 0; r < 8; ++r)
        soem[(rb + r) * H + col] = (_Float16)fmaxf(acc[r] + bb, 0.f);
    }
    __syncthreads();
    for (int nc = 0; nc < NC; ++nc) {
      v8f acc = {};
      for (int kc = 0; kc < 8; ++kc) {  // K 0..255 from other_em
        v16h a = load_a_lds(soem, H, kc * 32);
        v16h b = load_b_w(wv, 2 * H, nc * 16, kc * 32);
        acc = wmma16(a, b, acc);
      }
      for (int kc = 8; kc < 16; ++kc) {  // K 256..511 from self_em
        v16h a = load_a_lds(ssf, H, (kc - 8) * 32);
        v16h b = load_b_w(wv, 2 * H, nc * 16, kc * 32);
        acc = wmma16(a, b, acc);
      }
      const int col = nc * 16 + n15;
      const float bb = b_v[col];
#pragma unroll
      for (int r = 0; r < 8; ++r) {
        const int row = rb + r;
        const float vv = fmaxf(acc[r] + bb, 0.f);
        sh[row * H + col] += satt[row * 20 + j] * vv;  // lane-owned (row,col)
      }
    }
    __syncthreads();
  }

  // ---- h -> f16 (reuse soem) ----
  for (int i = lane; i < 16 * H; i += 32) soem[i] = (_Float16)sh[i];
  __syncthreads();

  // ---- out = relu([h, self_em] @ W_a + b_a) ----
  for (int nc = 0; nc < NC; ++nc) {
    v8f acc = {};
    for (int kc = 0; kc < 8; ++kc) {
      v16h a = load_a_lds(soem, H, kc * 32);
      v16h b = load_b_w(wa, 2 * H, nc * 16, kc * 32);
      acc = wmma16(a, b, acc);
    }
    for (int kc = 8; kc < 16; ++kc) {
      v16h a = load_a_lds(ssf, H, (kc - 8) * 32);
      v16h b = load_b_w(wa, 2 * H, nc * 16, kc * 32);
      acc = wmma16(a, b, acc);
    }
    const int col = nc * 16 + n15;
    const float bb = b_a[col];
#pragma unroll
    for (int r = 0; r < 8; ++r)
      out[(size_t)(row0 + rb + r) * H + col] = fmaxf(acc[r] + bb, 0.f);
  }
}

extern "C" void kernel_launch(void* const* d_in, const int* in_sizes, int n_in,
                              void* d_out, int out_size, void* d_ws,
                              size_t ws_size, hipStream_t stream) {
  const float* x      = (const float*)d_in[0];
  const float* eps    = (const float*)d_in[1];
  const float* W_mu   = (const float*)d_in[2];
  const float* b_mu   = (const float*)d_in[3];
  const float* W_var  = (const float*)d_in[4];
  const float* b_var  = (const float*)d_in[5];
  const float* W_self = (const float*)d_in[6];
  const float* b_self = (const float*)d_in[7];
  const float* W_oth  = (const float*)d_in[8];
  const float* b_oth  = (const float*)d_in[9];
  const float* W_q    = (const float*)d_in[10];
  const float* b_q    = (const float*)d_in[11];
  const float* W_k    = (const float*)d_in[12];
  const float* b_k    = (const float*)d_in[13];
  const float* W_v    = (const float*)d_in[14];
  const float* b_v    = (const float*)d_in[15];
  const float* W_a    = (const float*)d_in[16];
  const float* b_a    = (const float*)d_in[17];

  _Float16* wb = (_Float16*)d_ws;

  struct Job { const float* src; size_t off; int ko, kp; };
  const Job jobs[8] = {
      {W_mu, OFF_MU, OBS, KX},     {W_var, OFF_VAR, OBS, KX},
      {W_self, OFF_SELF, SD, 64},  {W_oth, OFF_OTH, OD, 32},
      {W_q, OFF_Q, H, H},          {W_k, OFF_K, H, H},
      {W_v, OFF_V, 2 * H, 2 * H},  {W_a, OFF_A, 2 * H, 2 * H}};
  for (int i = 0; i < 8; ++i) {
    const int total = H * jobs[i].kp;
    prep_w_kernel<<<(total + 255) / 256, 256, 0, stream>>>(
        jobs[i].src, wb + jobs[i].off, jobs[i].ko, jobs[i].kp);
  }

  // TDM path needs the padded f16 copy of x in ws (~20.3 MB total ws use).
  const int use_tdm = HAVE_TDM && (ws_size >= WS_HALVES_NEEDED * sizeof(_Float16));
  if (use_tdm) {
    const size_t total = (size_t)B_ * KX;
    prep_x_kernel<<<(unsigned)((total + 255) / 256), 256, 0, stream>>>(
        x, wb + OFF_X16);
  }

  // 512 blocks x 64 threads (2 waves); each wave owns a 16-row batch tile.
  fused_kernel<<<B_ / 32, 64, 2 * SLAB_BYTES, stream>>>(
      x, eps, b_mu, b_var, b_self, b_oth, b_q, b_k, b_v, b_a,
      wb + OFF_MU, wb + OFF_VAR, wb + OFF_SELF, wb + OFF_OTH, wb + OFF_Q,
      wb + OFF_K, wb + OFF_V, wb + OFF_A, wb + OFF_X16, use_tdm,
      (float*)d_out);
}
